// SurfaceAbstraction_18253611008387
// MI455X (gfx1250) — matrix operand order repaired and checked by reading, not compiled, not run on hardware
//
#include <hip/hip_runtime.h>
#include <hip/hip_bf16.h>
#include <math.h>

// ---------------- problem constants (match reference) ----------------
#define BB 4
#define NN 8192
#define SS 2048
#define KK 32
#define C_NORMAL 10
#define C_FEAT 32
#define IN_CH 48          // 3 + 3 + 10 + 32
#define IN_CH_PAD 64
#define RADIUS2 0.04f
#define EPS_BN 1e-5f
#define PP (BB * SS * KK) // 262144 points through the MLP
#define PI_F 3.14159265358979323846f

typedef _Float16 half_t;
typedef __attribute__((ext_vector_type(16))) _Float16 v16h;
typedef __attribute__((ext_vector_type(8)))  float    v8f;

// =====================================================================
// 1. Farthest point sampling: one workgroup per batch, dist[] in LDS.
//    Also writes new_center / new_normal outputs (B,3,S)/(B,10,S).
// =====================================================================
__global__ __launch_bounds__(1024)
void fps_kernel(const float* __restrict__ center,   // (B,3,N)
                const float* __restrict__ normal,   // (B,10,N)
                int* __restrict__ fpsIdx,           // (B,S)
                float* __restrict__ outCenter,      // (B,3,S)
                float* __restrict__ outNormal)      // (B,10,S)
{
    const int b = blockIdx.x;
    const float* cx = center + (long)b * 3 * NN;
    __shared__ float dist[NN];            // 32 KB
    __shared__ float redv[32];
    __shared__ int   redi[32];
    __shared__ int   sFar;

    const int t = threadIdx.x;
    for (int n = t; n < NN; n += 1024) dist[n] = 1e10f;
    if (t == 0) sFar = 0;
    __syncthreads();

    for (int s = 0; s < SS; ++s) {
        const int far = sFar;
        if (t == 0) {
            fpsIdx[b * SS + s] = far;
            for (int d = 0; d < 3; ++d)
                outCenter[(long)b * 3 * SS + d * SS + s] = cx[d * NN + far];
            for (int d = 0; d < C_NORMAL; ++d)
                outNormal[(long)b * C_NORMAL * SS + d * SS + s] =
                    normal[(long)b * C_NORMAL * NN + d * NN + far];
        }
        const float fx = cx[far], fy = cx[NN + far], fz = cx[2 * NN + far];

        float best = -1.0f; int bidx = 0;
        for (int n = t; n < NN; n += 1024) {
            float dx = cx[n] - fx, dy = cx[NN + n] - fy, dz = cx[2 * NN + n] - fz;
            float d  = dx * dx + dy * dy + dz * dz;
            float dv = fminf(dist[n], d);
            dist[n] = dv;
            if (dv > best || (dv == best && n < bidx)) { best = dv; bidx = n; }
        }
        // wave32 reduce (argmax, lowest index wins on ties like jnp.argmax)
        for (int off = 16; off; off >>= 1) {
            float ov = __shfl_down(best, off, 32);
            int   oi = __shfl_down(bidx, off, 32);
            if (ov > best || (ov == best && oi < bidx)) { best = ov; bidx = oi; }
        }
        const int wid = t >> 5, lane = t & 31;
        __syncthreads();
        if (lane == 0) { redv[wid] = best; redi[wid] = bidx; }
        __syncthreads();
        if (wid == 0) {
            best = redv[lane]; bidx = redi[lane];
            for (int off = 16; off; off >>= 1) {
                float ov = __shfl_down(best, off, 32);
                int   oi = __shfl_down(bidx, off, 32);
                if (ov > best || (ov == best && oi < bidx)) { best = ov; bidx = oi; }
            }
            if (lane == 0) sFar = bidx;
        }
        __syncthreads();
    }
}

// =====================================================================
// 2. Ball query + grouping + feature build. One wave per centroid.
//    Selects the first K indices (ascending) with d^2 <= r^2 — exactly
//    the reference's sort-then-slice semantics. Writes f16 activations
//    act0[p][c], p=((b*S+s)*K+k), c padded to 64 (48 real + 16 zeros).
// =====================================================================
__global__ __launch_bounds__(256)
void ball_group_kernel(const float* __restrict__ center,    // (B,3,N)
                       const float* __restrict__ normal,    // (B,10,N)
                       const float* __restrict__ feature,   // (B,32,N)
                       const float* __restrict__ newCenter, // (B,3,S) from d_out
                       half_t* __restrict__ act0)           // (P,64) f16
{
    const int gw   = blockIdx.x * (blockDim.x >> 5) + (threadIdx.x >> 5); // b*S+s
    const int lane = threadIdx.x & 31;
    const int b = gw / SS, s = gw % SS;
    const float* cx = center + (long)b * 3 * NN;

    const float qx = newCenter[(long)b * 3 * SS + 0 * SS + s];
    const float qy = newCenter[(long)b * 3 * SS + 1 * SS + s];
    const float qz = newCenter[(long)b * 3 * SS + 2 * SS + s];
    const float q2 = qx * qx + qy * qy + qz * qz;

    __shared__ int lists[8][KK];
    int* list = lists[threadIdx.x >> 5];

    int found = 0;
    for (int base = 0; base < NN && found < KK; base += 32) {
        const int n = base + lane;
        const float x = cx[n], y = cx[NN + n], z = cx[2 * NN + n];
        // expanded form, same as the reference
        const float sq = q2 + (x * x + y * y + z * z) - 2.0f * (qx * x + qy * y + qz * z);
        const bool in = !(sq > RADIUS2);
        const unsigned mask = (unsigned)__ballot(in);
        if (in) {
            const int pos = found + __popc(mask & ((1u << lane) - 1u));
            if (pos < KK) list[pos] = n;
        }
        found += __popc(mask);
    }
    __syncthreads();

    const int k   = lane;                       // K == 32 == wave size
    const int idx = (k < found) ? list[k] : list[0];

    const float gx = cx[idx], gy = cx[NN + idx], gz = cx[2 * NN + idx];
    const float dx = gx - qx, dy = gy - qy, dz = gz - qz;
    const float rho  = sqrtf(dx * dx + dy * dy + dz * dz);
    const float safe = (rho == 0.0f) ? 1.0f : rho;
    float theta = acosf(fminf(fmaxf(dz / safe, -1.0f), 1.0f));
    if (rho == 0.0f) theta = 0.0f;
    const float phi = atan2f(dy, dx);

    half_t* o = act0 + ((long)gw * KK + k) * IN_CH_PAD;
    o[0] = (half_t)dx;  o[1] = (half_t)dy;  o[2] = (half_t)dz;
    o[3] = (half_t)rho;
    o[4] = (half_t)(theta * (1.0f / PI_F));
    o[5] = (half_t)(phi * (1.0f / (2.0f * PI_F)) + 0.5f);
    for (int d = 0; d < C_NORMAL; ++d)
        o[6 + d] = (half_t)normal[(long)b * C_NORMAL * NN + d * NN + idx];
    for (int d = 0; d < C_FEAT; ++d)
        o[16 + d] = (half_t)feature[(long)b * C_FEAT * NN + d * NN + idx];
    for (int d = IN_CH; d < IN_CH_PAD; ++d) o[d] = (half_t)0.0f;
}

// =====================================================================
// 3a. Pack a layer's f32 weights (cout x cinReal) to padded f16
//     (cout x cinPad) so the WMMA kernel loads are branch-free.
// =====================================================================
__global__ void pack_weights_kernel(const float* __restrict__ W, half_t* __restrict__ Wp,
                                    int cout, int cinReal, int cinPad)
{
    int i = blockIdx.x * blockDim.x + threadIdx.x;
    if (i >= cout * cinPad) return;
    int m = i / cinPad, c = i % cinPad;
    Wp[i] = (c < cinReal) ? (half_t)W[m * cinReal + c] : (half_t)0.0f;
}

// =====================================================================
// 3b. WMMA GEMM: y[p][m] = sum_c Wp[m][c] * act[p][c] + bias[m]
//     One wave owns a 16(out-ch) x 16(point) D tile; K-loop in steps
//     of 32 via v_wmma_f32_16x16x32_f16. Grid is an exact multiple of
//     8 waves -> no divergence, EXEC all-1s at every WMMA.
// =====================================================================
__global__ __launch_bounds__(256)
void mlp_gemm_kernel(const half_t* __restrict__ act,  // (P, cinPad) f16
                     const half_t* __restrict__ Wp,   // (cout, cinPad) f16
                     const float*  __restrict__ bias, // (cout)
                     float* __restrict__ y,           // (P, cout) f32 pre-BN
                     int cinPad, int cout, int nTileCols /* cout/16 */)
{
    const int wave = blockIdx.x * (blockDim.x >> 5) + (threadIdx.x >> 5);
    const int lane = threadIdx.x & 31;
    const int  m0 = (wave % nTileCols) * 16;
    const long p0 = (long)(wave / nTileCols) * 16;
    const bool hi = lane >= 16;
    const int  lr = lane & 15;

    v8f acc = {};
    const half_t* arow = act + (p0 + lr) * cinPad + (hi ? 16 : 0);
    __builtin_prefetch(arow, 0, 1); // global_prefetch_b8

    for (int kc = 0; kc < cinPad; kc += 32) {
        // ---- A fragment: 16x32 f16, lane holds row m0+lr -------------
        v16h a;
        const half_t* wrow = Wp + (m0 + lr) * cinPad + kc + (hi ? 16 : 0);
#pragma unroll
        for (int g = 0; g < 8; ++g) {
            const int kb = 2 * g; // +hi offset folded into wrow
            a[2 * g]     = wrow[kb];
            a[2 * g + 1] = wrow[kb + 1];
        }
        // ---- B fragment: 32x16 f16, lane holds col p0+lr, 16 contig K
        v16h bf;
        const half_t* brow = arow + kc;
#pragma unroll
        for (int e = 0; e < 16; ++e) bf[e] = brow[e];

        acc = __builtin_amdgcn_wmma_f32_16x16x32_f16(
            /*neg_a=*/false, a, /*neg_b=*/false, bf,
            /*c_mod=*/(short)0, acc, /*reuse_a=*/false, /*reuse_b=*/false);
    }

    // D layout: element r -> m = m0 + r + (hi?8:0), point = p0 + lr
#pragma unroll
    for (int r = 0; r < 8; ++r) {
        const int m = m0 + r + (hi ? 8 : 0);
        y[(p0 + lr) * cout + m] = acc[r] + bias[m];
    }
}

// =====================================================================
// 3c. Per-channel BN stats over all P points (deterministic tree sum).
// =====================================================================
__global__ __launch_bounds__(256)
void bn_stats_kernel(const float* __restrict__ y, int cout,
                     float* __restrict__ mu, float* __restrict__ rsig)
{
    const int c = blockIdx.x;
    const int t = threadIdx.x;
    float s = 0.0f, s2 = 0.0f;
    for (long p = t; p < (long)PP; p += 256) {
        const float v = y[p * cout + c];
        s += v; s2 += v * v;
    }
    __shared__ float ss[256], ss2[256];
    ss[t] = s; ss2[t] = s2; __syncthreads();
    for (int o = 128; o; o >>= 1) {
        if (t < o) { ss[t] += ss[t + o]; ss2[t] += ss2[t + o]; }
        __syncthreads();
    }
    if (t == 0) {
        const float inv = 1.0f / (float)PP;
        const float m   = ss[0] * inv;
        const float var = ss2[0] * inv - m * m;
        mu[c] = m;
        rsig[c] = rsqrtf(var + EPS_BN);
    }
}

// =====================================================================
// 3d. BN + affine + ReLU -> f16 activations for the next layer.
// =====================================================================
__global__ __launch_bounds__(256)
void bn_relu_f16_kernel(const float* __restrict__ y,
                        const float* __restrict__ mu, const float* __restrict__ rsig,
                        const float* __restrict__ g,  const float* __restrict__ beta,
                        int cout, half_t* __restrict__ out)
{
    const long i = (long)blockIdx.x * blockDim.x + threadIdx.x;
    if (i >= (long)PP * cout) return;
    const int c = (int)(i % cout);
    const float v = (y[i] - mu[c]) * rsig[c] * g[c] + beta[c];
    out[i] = (half_t)fmaxf(v, 0.0f);
}

// =====================================================================
// 4. Final layer: BN + affine + ReLU fused with max over K.
//    outFeat is (B,128,S).
// =====================================================================
__global__ __launch_bounds__(256)
void bn_relu_max_kernel(const float* __restrict__ y2,
                        const float* __restrict__ mu, const float* __restrict__ rsig,
                        const float* __restrict__ g,  const float* __restrict__ beta,
                        int cout, float* __restrict__ outFeat)
{
    const long i = (long)blockIdx.x * blockDim.x + threadIdx.x;
    if (i >= (long)BB * SS * cout) return;
    const int  c  = (int)(i % cout);
    const long bs = i / cout;
    const int  s  = (int)(bs % SS);
    const int  b  = (int)(bs / SS);

    const float rs = rsig[c], m = mu[c], gg = g[c], be = beta[c];
    float best = 0.0f; // ReLU outputs are >= 0
    const float* yp = y2 + ((long)(b * SS + s) * KK) * cout + c;
    for (int k = 0; k < KK; ++k) {
        const float v = fmaxf(gg * ((yp[(long)k * cout] - m) * rs) + be, 0.0f);
        best = fmaxf(best, v);
    }
    outFeat[(long)b * cout * SS + (long)c * SS + s] = best;
}

// =====================================================================
// launch
// =====================================================================
extern "C" void kernel_launch(void* const* d_in, const int* in_sizes, int n_in,
                              void* d_out, int out_size, void* d_ws, size_t ws_size,
                              hipStream_t stream)
{
    const float* center  = (const float*)d_in[0];
    const float* normal  = (const float*)d_in[1];
    const float* feature = (const float*)d_in[2];
    const float* Wl[3]    = { (const float*)d_in[3],  (const float*)d_in[7],  (const float*)d_in[11] };
    const float* Bl[3]    = { (const float*)d_in[4],  (const float*)d_in[8],  (const float*)d_in[12] };
    const float* Gl[3]    = { (const float*)d_in[5],  (const float*)d_in[9],  (const float*)d_in[13] };
    const float* BeL[3]   = { (const float*)d_in[6],  (const float*)d_in[10], (const float*)d_in[14] };
    const int cinR[3]  = { IN_CH, 64, 64 };
    const int cinP[3]  = { IN_CH_PAD, 64, 64 };
    const int coutL[3] = { 64, 64, 128 };

    float* out       = (float*)d_out;
    float* outCenter = out;                                  // (B,3,S)
    float* outNormal = out + (long)BB * 3 * SS;              // (B,10,S)
    float* outFeat   = out + (long)BB * 3 * SS + (long)BB * C_NORMAL * SS; // (B,128,S)

    // ---- workspace carve ------------------------------------------------
    char* w = (char*)d_ws;
    auto carve = [&](size_t bytes) { char* p = w; w += (bytes + 255) & ~(size_t)255; return p; };
    int*    fpsIdx = (int*)   carve((size_t)BB * SS * sizeof(int));
    float*  mu     = (float*) carve(128 * sizeof(float));
    float*  rsig   = (float*) carve(128 * sizeof(float));
    half_t* Wp[3];
    for (int l = 0; l < 3; ++l)
        Wp[l] = (half_t*)carve((size_t)coutL[l] * cinP[l] * sizeof(half_t));
    float*  y    = (float*) carve((size_t)PP * 128 * sizeof(float));      // 128 MB
    half_t* actA = (half_t*)carve((size_t)PP * 64 * sizeof(half_t));      // 32 MB
    half_t* actB = (half_t*)carve((size_t)PP * 64 * sizeof(half_t));      // 32 MB
    (void)ws_size; (void)n_in; (void)in_sizes; (void)out_size;

    // ---- 1. FPS (writes new_center / new_normal outputs too) ------------
    fps_kernel<<<BB, 1024, 0, stream>>>(center, normal, fpsIdx, outCenter, outNormal);

    // ---- 2. ball query + grouping -> actA (P x 64 f16) ------------------
    ball_group_kernel<<<(BB * SS) / 8, 256, 0, stream>>>(center, normal, feature,
                                                         outCenter, actA);

    // ---- 3. pack weights ------------------------------------------------
    for (int l = 0; l < 3; ++l) {
        const int n = coutL[l] * cinP[l];
        pack_weights_kernel<<<(n + 255) / 256, 256, 0, stream>>>(Wl[l], Wp[l],
                                                                 coutL[l], cinR[l], cinP[l]);
    }

    // ---- 4. MLP layers --------------------------------------------------
    const half_t* actIn = actA;
    half_t*       actOut = actB;
    for (int l = 0; l < 3; ++l) {
        const int nTileCols = coutL[l] / 16;
        const int totalWaves = (PP / 16) * nTileCols;  // multiple of 8
        mlp_gemm_kernel<<<totalWaves / 8, 256, 0, stream>>>(actIn, Wp[l], Bl[l], y,
                                                            cinP[l], coutL[l], nTileCols);
        bn_stats_kernel<<<coutL[l], 256, 0, stream>>>(y, coutL[l], mu, rsig);
        if (l < 2) {
            const long tot = (long)PP * coutL[l];
            bn_relu_f16_kernel<<<(int)((tot + 255) / 256), 256, 0, stream>>>(
                y, mu, rsig, Gl[l], BeL[l], coutL[l], actOut);
            actIn = actOut;
            actOut = (actOut == actB) ? actA : actB;
        } else {
            const long tot = (long)BB * SS * coutL[l];
            bn_relu_max_kernel<<<(int)((tot + 255) / 256), 256, 0, stream>>>(
                y, mu, rsig, Gl[l], BeL[l], coutL[l], outFeat);
        }
    }
}
